// KNN_Embedding_V_15960098472027
// MI455X (gfx1250) — matrix operand order repaired and checked by reading, not compiled, not run on hardware
//
#include <hip/hip_runtime.h>
#include <hip/hip_bf16.h>

typedef __attribute__((ext_vector_type(16))) __bf16 v16bf;
typedef __attribute__((ext_vector_type(8)))  __bf16 v8bf;
typedef __attribute__((ext_vector_type(8)))  float  v8f;

#define B_   8
#define N_   4096
#define D_   128
#define E_   512
#define K_   27
#define FAN  (D_*K_)        // 3456
#define KSTEPS (FAN/32)     // 108
#define NTILES (E_/16)      // 32
#define TOPK_CHUNK 512

__device__ __forceinline__ unsigned short f2bf(float f){
  unsigned u = __float_as_uint(f);
  u += 0x7fffu + ((u >> 16) & 1u);          // round-to-nearest-even
  return (unsigned short)(u >> 16);
}

// ---------------- Kernel 1: x (f32) -> bf16 ----------------
__global__ void k_cvt_x(const float* __restrict__ x, unsigned short* __restrict__ xbf, int n){
  int i = blockIdx.x * blockDim.x + threadIdx.x;
  if (i < n) xbf[i] = f2bf(x[i]);
}

// ---------------- Kernel 2: pack W into WMMA B-fragment layout ----------------
// Wp index = ((ntile*KSTEPS + kstep)*32 + lane)*16 + j
// element  = W[e = ntile*16 + lane%16][f = kstep*32 + (lane<16?0:16) + j]
__global__ void k_pack_w(const float* __restrict__ W, unsigned short* __restrict__ Wp, int n){
  int t = blockIdx.x * blockDim.x + threadIdx.x;
  if (t >= n) return;
  int j     = t & 15;
  int lane  = (t >> 4) & 31;
  int kstep = (t >> 9) % KSTEPS;
  int ntile = (t >> 9) / KSTEPS;
  int e = ntile*16 + (lane & 15);
  int f = kstep*32 + ((lane >> 4) << 4) + j;
  Wp[t] = f2bf(W[(size_t)e * FAN + f]);
}

// ---------------- Kernel 3: top-27 nearest neighbors, sorted ascending ----------------
// one wave32 per query; 8 queries (waves) per block; candidates staged in LDS
__global__ __launch_bounds__(256) void k_topk(const float* __restrict__ xv, int* __restrict__ outIdx){
  __shared__ float s_chunk[TOPK_CHUNK*3];   //  6 KB
  __shared__ float s_md[8*32*K_];           // 27 KB
  __shared__ int   s_mi[8*32*K_];           // 27 KB
  const int tid  = threadIdx.x;
  const int wave = tid >> 5, lane = tid & 31;
  const int q    = blockIdx.x * 8 + wave;   // global query row (b*N + n)
  const int b    = q >> 12;

  float qx = xv[(size_t)q*3 + 0];
  float qy = xv[(size_t)q*3 + 1];
  float qz = xv[(size_t)q*3 + 2];

  float bd[K_]; int bi[K_];
  #pragma unroll
  for (int j = 0; j < K_; ++j){ bd[j] = 3.4e38f; bi[j] = -1; }

  const size_t candBase = (size_t)b * N_ * 3;
  for (int c = 0; c < N_/TOPK_CHUNK; ++c){
    __syncthreads();
    for (int t = tid; t < TOPK_CHUNK*3; t += 256)
      s_chunk[t] = xv[candBase + (size_t)c*TOPK_CHUNK*3 + t];
    __syncthreads();
    for (int i = lane; i < TOPK_CHUNK; i += 32){
      float dx = s_chunk[i*3+0] - qx;
      float dy = s_chunk[i*3+1] - qy;
      float dz = s_chunk[i*3+2] - qz;
      float d  = dx*dx + dy*dy + dz*dz;
      int   m  = c*TOPK_CHUNK + i;
      if (d < bd[K_-1]){
        bd[K_-1] = d; bi[K_-1] = m;
        // single unrolled bubble pass keeps the register list sorted ascending
        #pragma unroll
        for (int j = K_-2; j >= 0; --j){
          if (bd[j] > bd[j+1]){
            float td = bd[j]; bd[j] = bd[j+1]; bd[j+1] = td;
            int   ti = bi[j]; bi[j] = bi[j+1]; bi[j+1] = ti;
          }
        }
      }
    }
  }

  // publish per-lane sorted lists, then 27-round 32-way merge
  const int base = wave * 32 * K_;
  #pragma unroll
  for (int j = 0; j < K_; ++j){
    s_md[base + lane*K_ + j] = bd[j];
    s_mi[base + lane*K_ + j] = bi[j];
  }
  __syncthreads();

  int p = 0;
  for (int r = 0; r < K_; ++r){
    float d  = (p < K_) ? s_md[base + lane*K_ + p] : 3.4e38f;
    int   ai = (p < K_) ? s_mi[base + lane*K_ + p] : -1;
    int   src = lane;
    #pragma unroll
    for (int off = 16; off > 0; off >>= 1){
      float od = __shfl_xor(d,   off, 32);
      int   oi = __shfl_xor(ai,  off, 32);
      int   os = __shfl_xor(src, off, 32);
      bool take = (od < d) || (od == d && os < src);
      d = take ? od : d; ai = take ? oi : ai; src = take ? os : src;
    }
    if (lane == src) p++;
    if (lane == 0) outIdx[(size_t)q*K_ + r] = ai;
  }
}

// ---------------- Kernel 4: gather + bf16 WMMA GEMM + bias ----------------
// block: 32 query rows (two M-tiles), 8 waves; wave owns 4 N-tiles (64 cols),
// reusing each B fragment for both M-tiles; B double-buffered across sub-steps.
// waves_per_eu(4): allow ~256 VGPRs so all 4 prefetched B fragments stay live.
__global__ __launch_bounds__(256)
__attribute__((amdgpu_waves_per_eu(4)))
void k_gemm(const unsigned short* __restrict__ xbf,
            const unsigned short* __restrict__ Wp,
            const int*   __restrict__ nbr,
            const float* __restrict__ bias,
            float* __restrict__ out){
  __shared__ unsigned short sA[32*136];     // 2 tiles x 16 rows x 128 bf16, padded stride
  __shared__ int sIdx[32*K_];
  const int tid     = threadIdx.x;
  const int wave    = tid >> 5, lane = tid & 31;
  const int rowBase = blockIdx.x * 32;      // global row (b*N + n), never crosses batch
  const int b       = rowBase >> 12;

  for (int t = tid; t < 32*K_; t += 256)
    sIdx[t] = nbr[(size_t)(rowBase + t/K_)*K_ + (t % K_)];

  v8f acc[2][4];
  #pragma unroll
  for (int mt = 0; mt < 2; ++mt)
    #pragma unroll
    for (int nt = 0; nt < 4; ++nt)
      #pragma unroll
      for (int r = 0; r < 8; ++r) acc[mt][nt][r] = 0.0f;

  const int kBase = (lane < 16) ? 0 : 16;   // A-fragment K base, contiguous k = 16h + s (matches packed B)
  const int mrow  = lane & 15;              // A-fragment M row per ISA layout
  const int aOff0 = mrow*136 + kBase;       // LDS element base, M-tile 0
  const int aOff1 = aOff0 + 16*136;         // LDS element base, M-tile 1

  // single per-lane B base pointer; all fragment loads use constant immediate
  // offsets (nt*KSTEPS*1024B + sub*1024B); advanced by 4 KB per k0.
  const unsigned short* bptr = Wp + ((((size_t)wave*4*KSTEPS)*32 + lane) << 4);

  for (int k0 = 0; k0 < K_; ++k0){
    // issue sub-0 B fragment loads early; they overlap the A gather below
    v16bf bcur[4];
    #pragma unroll
    for (int nt = 0; nt < 4; ++nt)
      bcur[nt] = *(const v16bf*)(bptr + nt*(KSTEPS*512));

    __syncthreads();
    for (int t = tid; t < 512; t += 256){   // 32 rows x 16 segments of 16 B
      int row = t >> 4, seg = t & 15;
      int nidx = sIdx[row*K_ + k0];
      const uint4* src = (const uint4*)(xbf + ((size_t)(b*N_ + nidx)*D_ + seg*8));
      *(uint4*)(&sA[row*136 + seg*8]) = *src;
    }
    __syncthreads();

    #pragma unroll
    for (int sub = 0; sub < 4; ++sub){
      v16bf bnext[4];
      if (sub < 3){
        #pragma unroll
        for (int nt = 0; nt < 4; ++nt)
          bnext[nt] = *(const v16bf*)(bptr + nt*(KSTEPS*512) + (sub+1)*512);
      }
      v16bf a0, a1;
      {
        v8bf lo = *(const v8bf*)((const __bf16*)sA + aOff0 + sub*32);
        v8bf hi = *(const v8bf*)((const __bf16*)sA + aOff0 + sub*32 + 8);
        #pragma unroll
        for (int j = 0; j < 8; ++j){ a0[j] = lo[j]; a0[j+8] = hi[j]; }
      }
      {
        v8bf lo = *(const v8bf*)((const __bf16*)sA + aOff1 + sub*32);
        v8bf hi = *(const v8bf*)((const __bf16*)sA + aOff1 + sub*32 + 8);
        #pragma unroll
        for (int j = 0; j < 8; ++j){ a1[j] = lo[j]; a1[j+8] = hi[j]; }
      }
      // keep the prefetch loads above, the matrix math below
      __builtin_amdgcn_sched_barrier(0);
      #pragma unroll
      for (int nt = 0; nt < 4; ++nt){
        acc[0][nt] = __builtin_amdgcn_wmma_f32_16x16x32_bf16(
            false, a0, false, bcur[nt], (short)0, acc[0][nt], false, false);
        acc[1][nt] = __builtin_amdgcn_wmma_f32_16x16x32_bf16(
            false, a1, false, bcur[nt], (short)0, acc[1][nt], false, false);
      }
      if (sub < 3){
        #pragma unroll
        for (int nt = 0; nt < 4; ++nt) bcur[nt] = bnext[nt];
      }
    }
    bptr += 2048;                            // 4 ksteps * 512 elements
  }

  // epilogue: C layout — VGPR r holds M = r + (lane<16?0:8), N = lane%16
  const int col  = lane & 15;
  const int rOff = (lane >> 4) << 3;
  #pragma unroll
  for (int nt = 0; nt < 4; ++nt){
    const int e = (wave*4 + nt)*16 + col;
    const float bv = bias[e];
    #pragma unroll
    for (int mt = 0; mt < 2; ++mt){
      #pragma unroll
      for (int r = 0; r < 8; ++r){
        const int g = rowBase + mt*16 + rOff + r;
        out[(size_t)g*E_ + e] = acc[mt][nt][r] + bv;
      }
    }
  }
}

extern "C" void kernel_launch(void* const* d_in, const int* in_sizes, int n_in,
                              void* d_out, int out_size, void* d_ws, size_t ws_size,
                              hipStream_t stream){
  const float* x    = (const float*)d_in[0];   // [B,N,D]
  const float* xv   = (const float*)d_in[1];   // [B,N,3]
  const float* W    = (const float*)d_in[2];   // [E,FAN]
  const float* bias = (const float*)d_in[3];   // [E]
  float* out = (float*)d_out;                  // [B,N,E] f32

  char* ws = (char*)d_ws;
  unsigned short* xbf = (unsigned short*)(ws);                                   // 8,388,608 B
  unsigned short* Wp  = (unsigned short*)(ws + (size_t)B_*N_*D_*2);              // 3,538,944 B
  int*            nbr = (int*)(ws + (size_t)B_*N_*D_*2 + (size_t)E_*FAN*2);      // 3,538,944 B

  const int nX = B_*N_*D_;
  hipLaunchKernelGGL(k_cvt_x, dim3((nX + 255)/256), dim3(256), 0, stream, x, xbf, nX);
  const int nW = E_*FAN;
  hipLaunchKernelGGL(k_pack_w, dim3((nW + 255)/256), dim3(256), 0, stream, W, Wp, nW);
  hipLaunchKernelGGL(k_topk, dim3(B_*N_/8), dim3(256), 0, stream, xv, nbr);
  hipLaunchKernelGGL(k_gemm, dim3(B_*N_/32), dim3(256), 0, stream, xbf, Wp, nbr, bias, out);
}
